// TorchVectorSimilarity_16527034155636
// MI455X (gfx1250) — matrix operand order repaired and checked by reading, not compiled, not run on hardware
//
#include <hip/hip_runtime.h>
#include <hip/hip_bf16.h>

typedef __attribute__((ext_vector_type(16))) __bf16 bvec16;
typedef __attribute__((ext_vector_type(8)))  float  fvec8;

#define QB     128      // number of query rows (B)
#define DIM    128      // vector dimension (D == K of the GEMM)
#define KTOP   10       // top-k
#define CHUNKS 8        // partial top-k chunks per row
#define NTILE  4        // 16-col N-subtiles per block (64 db vectors/block)
#define KSTEPS (DIM / 32)
#define NEGINF (-3.402823e38f)

// ---------------------------------------------------------------------------
// Kernel 0: normalize queries ONCE -> bf16 buffer in workspace (L2-resident).
// 1 block, 256 threads: 2 threads per row, 64 elements each, kept in regs.
// ---------------------------------------------------------------------------
__global__ __launch_bounds__(256) void normalize_q(const float* __restrict__ q,
                                                   __bf16* __restrict__ qbf) {
  __shared__ float qss[QB];
  const int tid = threadIdx.x;
  const int row = tid >> 1, half = tid & 1;

  if (tid < QB) qss[tid] = 0.f;
  __syncthreads();

  float v[64];
  const float* s = q + row * DIM + half * 64;
  float acc = 0.f;
#pragma unroll 8
  for (int i = 0; i < 64; ++i) { v[i] = s[i]; acc += v[i] * v[i]; }
  atomicAdd(&qss[row], acc);
  __syncthreads();

  const float inv = 1.0f / fmaxf(sqrtf(qss[row]), 1e-12f);
  __bf16* dst = qbf + row * DIM + half * 64;
#pragma unroll 8
  for (int i = 0; i < 64; ++i) dst[i] = (__bf16)(v[i] * inv);
}

// ---------------------------------------------------------------------------
// Kernel 1: fused db-normalize + WMMA GEMM.
// One block = 256 threads = 8 waves handles 64 db vectors (4 N-subtiles)
// across all 128 query rows. db tile normalized in a SINGLE global pass
// (values in registers, ds_add_f32 reduction), converted to bf16 in LDS.
// WMMA phase: A fragments (full K=128) loaded once from the precomputed bf16
// Q buffer; B fragments double-buffered in registers so each K-step's 4
// WMMAs overlap the next K-step's 8 ds_load_b128s.
// 16 x v_wmma_f32_16x16x32_bf16 per wave.
// ---------------------------------------------------------------------------
__global__ __launch_bounds__(256) void sim_gemm(const __bf16* __restrict__ qbf,
                                                const float* __restrict__ db,
                                                float* __restrict__ sims,
                                                long long Ncols) {
  __shared__ __bf16 dbs[16 * NTILE][DIM];   // 16 KB: normalized db tile, bf16
  __shared__ float  dss[16 * NTILE];

  const int tid = threadIdx.x;
  const long long colBase = (long long)blockIdx.x * (16 * NTILE);

  if (tid < 16 * NTILE) dss[tid] = 0.f;
  __syncthreads();

  // ---- db tile: single-pass normalize. 4 threads/vector, 32 dims each.
  {
    const int vec = tid >> 2, c = tid & 3;      // vec 0..63, quarter 0..3
    const float* s = db + (colBase + vec) * DIM + c * 32;
    float v[32];
    float acc = 0.f;
#pragma unroll 8
    for (int i = 0; i < 32; ++i) { v[i] = s[i]; acc += v[i] * v[i]; }
    atomicAdd(&dss[vec], acc);
    __builtin_prefetch(db + (colBase + 16 * NTILE) * DIM, 0, 1);
    __syncthreads();
    const float inv = 1.0f / fmaxf(sqrtf(dss[vec]), 1e-12f);
    __bf16* dst = &dbs[vec][c * 32];
#pragma unroll 8
    for (int i = 0; i < 32; ++i) dst[i] = (__bf16)(v[i] * inv);
  }
  __syncthreads();

  // ---- WMMA phase
  const int wave   = tid >> 5;   // 0..7 -> M tile
  const int lane   = tid & 31;
  const int half16 = lane >> 4;  // lane half selects K sub-range (ISA 7.12.2)
  const int lane16 = lane & 15;

  union Frag { bvec16 v; uint4 u[2]; };

  // A fragments for all K-steps, loaded once from L2-resident bf16 Q.
  // Lane holds row M=lane16 of M-tile `wave`; lanes 0-15: K [kb..kb+15],
  // lanes 16-31: K [kb+16..kb+31].
  Frag Af[KSTEPS];
  const __bf16* arow = qbf + (wave * 16 + lane16) * DIM;
#pragma unroll
  for (int k = 0; k < KSTEPS; ++k) {
    const int kb = k * 32;
    Af[k].u[0] = *(const uint4*)(arow + kb + half16 * 16);
    Af[k].u[1] = *(const uint4*)(arow + kb + half16 * 16 + 8);
  }

  fvec8 acc[NTILE];
#pragma unroll
  for (int n = 0; n < NTILE; ++n)
    acc[n] = (fvec8){0.f, 0.f, 0.f, 0.f, 0.f, 0.f, 0.f, 0.f};

  // B fragments, double-buffered in registers. Lane holds column N=lane16
  // of subtile n; lanes 0-15: K [kb..kb+15], lanes 16-31: K [kb+16..kb+31].
  Frag Bf[2][NTILE];
#pragma unroll
  for (int n = 0; n < NTILE; ++n) {
    const __bf16* bcol = &dbs[n * 16 + lane16][half16 * 16];
    Bf[0][n].u[0] = *(const uint4*)(bcol);
    Bf[0][n].u[1] = *(const uint4*)(bcol + 8);
  }

#pragma unroll
  for (int k = 0; k < KSTEPS; ++k) {
    const int cur = k & 1, nxt = cur ^ 1;
    if (k + 1 < KSTEPS) {
      const int kb = (k + 1) * 32;
#pragma unroll
      for (int n = 0; n < NTILE; ++n) {
        const __bf16* bcol = &dbs[n * 16 + lane16][kb + half16 * 16];
        Bf[nxt][n].u[0] = *(const uint4*)(bcol);
        Bf[nxt][n].u[1] = *(const uint4*)(bcol + 8);
      }
    }
#pragma unroll
    for (int n = 0; n < NTILE; ++n) {
      acc[n] = __builtin_amdgcn_wmma_f32_16x16x32_bf16(
          /*neg_a=*/false, Af[k].v, /*neg_b=*/false, Bf[cur][n].v,
          /*c_mod=*/(short)0, acc[n], /*reuse_a=*/false, /*reuse_b=*/false);
    }
  }

  // C layout: VGPR i = (M=i, N=lane) lanes 0-15; (M=8+i) lanes 16-31.
#pragma unroll
  for (int n = 0; n < NTILE; ++n) {
    const long long col = colBase + n * 16 + lane16;
#pragma unroll
    for (int i = 0; i < 8; ++i) {
      const int row = wave * 16 + i + half16 * 8;
      sims[(long long)row * Ncols + col] = acc[n][i];
    }
  }
}

// ---------------------------------------------------------------------------
// Kernel 2: partial top-k. grid = QB*CHUNKS blocks; per-thread register
// top-10 (unrolled compare-swap => pure VALU), LDS tree merge -> ws.
// ---------------------------------------------------------------------------
__global__ __launch_bounds__(256) void topk_partial(const float* __restrict__ sims,
                                                    float* __restrict__ wsV,
                                                    int* __restrict__ wsI,
                                                    long long Ncols) {
  __shared__ float sv[256 * KTOP];
  __shared__ int   si[256 * KTOP];

  const int tid   = threadIdx.x;
  const int qrow  = blockIdx.x / CHUNKS;
  const int chunk = blockIdx.x % CHUNKS;
  const long long chunkLen = (Ncols + CHUNKS - 1) / CHUNKS;
  const long long start = (long long)chunk * chunkLen;
  long long end = start + chunkLen;
  if (end > Ncols) end = Ncols;

  float bv[KTOP]; int bi[KTOP];
#pragma unroll
  for (int j = 0; j < KTOP; ++j) { bv[j] = NEGINF; bi[j] = 0; }

  const float* rowp = sims + (long long)qrow * Ncols;
  for (long long c = start + tid; c < end; c += 256) {
    float v = rowp[c];
    if (v > bv[KTOP - 1]) {
      int ix = (int)c;
#pragma unroll
      for (int j = 0; j < KTOP; ++j) {
        if (v > bv[j]) {
          float tv = bv[j]; bv[j] = v;  v  = tv;
          int   ti = bi[j]; bi[j] = ix; ix = ti;
        }
      }
    }
  }

#pragma unroll
  for (int j = 0; j < KTOP; ++j) { sv[tid * KTOP + j] = bv[j]; si[tid * KTOP + j] = bi[j]; }

  for (int s = 128; s > 0; s >>= 1) {
    __syncthreads();
    if (tid < s) {
#pragma unroll
      for (int j = 0; j < KTOP; ++j) {
        float v  = sv[(tid + s) * KTOP + j];
        int   ix = si[(tid + s) * KTOP + j];
#pragma unroll
        for (int m = 0; m < KTOP; ++m) {
          if (v > sv[tid * KTOP + m]) {
            float tv = sv[tid * KTOP + m]; sv[tid * KTOP + m] = v;  v  = tv;
            int   ti = si[tid * KTOP + m]; si[tid * KTOP + m] = ix; ix = ti;
          }
        }
      }
    }
  }
  __syncthreads();

  if (tid == 0) {
#pragma unroll
    for (int j = 0; j < KTOP; ++j) {
      wsV[(long long)blockIdx.x * KTOP + j] = sv[j];
      wsI[(long long)blockIdx.x * KTOP + j] = si[j];
    }
  }
}

// ---------------------------------------------------------------------------
// Kernel 3: final merge. 1 block, QB threads; thread q merges its CHUNKS
// partial lists and writes indices (as float, exact for idx < 2^24).
// ---------------------------------------------------------------------------
__global__ void topk_final(const float* __restrict__ wsV,
                           const int* __restrict__ wsI,
                           float* __restrict__ outIdx) {
  const int qrow = threadIdx.x;
  float bv[KTOP]; int bi[KTOP];
#pragma unroll
  for (int j = 0; j < KTOP; ++j) { bv[j] = NEGINF; bi[j] = 0; }

  for (int c = 0; c < CHUNKS; ++c) {
#pragma unroll
    for (int j = 0; j < KTOP; ++j) {
      float v  = wsV[((long long)qrow * CHUNKS + c) * KTOP + j];
      int   ix = wsI[((long long)qrow * CHUNKS + c) * KTOP + j];
      if (v > bv[KTOP - 1]) {
#pragma unroll
        for (int m = 0; m < KTOP; ++m) {
          if (v > bv[m]) {
            float tv = bv[m]; bv[m] = v;  v  = tv;
            int   ti = bi[m]; bi[m] = ix; ix = ti;
          }
        }
      }
    }
  }
#pragma unroll
  for (int j = 0; j < KTOP; ++j) outIdx[qrow * KTOP + j] = (float)bi[j];
}

// ---------------------------------------------------------------------------
extern "C" void kernel_launch(void* const* d_in, const int* in_sizes, int n_in,
                              void* d_out, int out_size, void* d_ws, size_t ws_size,
                              hipStream_t stream) {
  const float* q  = (const float*)d_in[0];   // [B, D] fp32
  const float* db = (const float*)d_in[1];   // [N, D] fp32
  const int Bq = in_sizes[0] / DIM;                    // 128
  const long long N = (long long)in_sizes[1] / DIM;    // 1,000,000

  // Output layout (tuple flattened): [B*KTOP] indices, then [B*N] similarities
  float* outIdx = (float*)d_out;
  float* sims   = (float*)d_out + (size_t)Bq * KTOP;

  // Workspace layout: partial top-k values | partial top-k indices | bf16 Q
  const size_t partials = (size_t)Bq * CHUNKS * KTOP;
  float*  wsV = (float*)d_ws;
  int*    wsI = (int*)((char*)d_ws + partials * sizeof(float));
  __bf16* qbf = (__bf16*)((char*)d_ws + partials * (sizeof(float) + sizeof(int)));

  normalize_q<<<dim3(1), 256, 0, stream>>>(q, qbf);
  sim_gemm<<<dim3((unsigned)(N / (16 * NTILE))), 256, 0, stream>>>(qbf, db, sims, N);
  topk_partial<<<dim3((unsigned)(Bq * CHUNKS)), 256, 0, stream>>>(sims, wsV, wsI, N);
  topk_final<<<dim3(1), dim3((unsigned)Bq), 0, stream>>>(wsV, wsI, outIdx);
}
